// DualPathAttention_50036368998670
// MI455X (gfx1250) — matrix operand order repaired and checked by reading, not compiled, not run on hardware
//
#include <hip/hip_runtime.h>

// ---------------------------------------------------------------- types
typedef unsigned short u16;
typedef __bf16 bf16;
typedef bf16  bf16x16 __attribute__((ext_vector_type(16)));
typedef float v8f     __attribute__((ext_vector_type(8)));
typedef unsigned int  u32x4 __attribute__((ext_vector_type(4)));
typedef u16           u16x4 __attribute__((ext_vector_type(4)));

union Frag16 {          // one WMMA bf16 operand (16 halves = 8 VGPRs / lane)
  bf16x16 bf;
  u16     u[16];
  u32x4   q[2];
};

// CDNA5 async global->LDS path (probe-guarded; falls back to VGPR bounce)
#if __has_builtin(__builtin_amdgcn_global_load_async_to_lds_b128)
#define HAVE_ASYNC_LDS 1
typedef int v4i __attribute__((vector_size(16)));                // matches builtin proto
typedef __attribute__((address_space(1))) v4i gv4i;              // global int4*
typedef __attribute__((address_space(3))) v4i sv4i;              // LDS int4*
#else
#define HAVE_ASYNC_LDS 0
#endif

__device__ __forceinline__ void async_wait0() {
#if HAVE_ASYNC_LDS
#if __has_builtin(__builtin_amdgcn_s_wait_asynccnt)
  __builtin_amdgcn_s_wait_asynccnt(0);
#else
  asm volatile("s_wait_asynccnt 0x0" ::: "memory");
#endif
#else
  asm volatile("s_wait_dscnt 0x0" ::: "memory");
#endif
}

// problem constants
constexpr int Bn = 32, Sn = 512, Dn = 512;
constexpr int GHn = 8, LHn = 4;
constexpr int Wn = 10, STRIDEn = 5, NWn = 103;           // ceil(512/5)=103 windows
constexpr int NLOC = Bn * NWn * Wn;                      // 32960 local rows

__device__ __forceinline__ u16 f2bf(float f) {           // f32 -> bf16 RNE
  unsigned int u = __builtin_bit_cast(unsigned int, f);
  unsigned int r = (u + 0x7FFFu + ((u >> 16) & 1u)) >> 16;
  return (u16)r;
}

__device__ __forceinline__ v8f wmma_bf16(const Frag16& a, const Frag16& b, v8f c) {
  return __builtin_amdgcn_wmma_f32_16x16x32_bf16(false, a.bf, false, b.bf,
                                                 (short)0, c, false, false);
}

// local-window logical row -> physical row of x  (b*512 + clamp(5*nw + w))
__device__ __forceinline__ size_t map_row_local(int r) {
  int b   = r / (NWn * Wn);
  int rem = r - b * (NWn * Wn);
  int nw  = rem / Wn;
  int w   = rem - nw * Wn;
  int pos = nw * STRIDEn + w;
  if (pos > Sn - 1) pos = Sn - 1;
  return (size_t)b * Sn + pos;
}

// ---------------------------------------------------------------- f32 -> bf16 (x4)
__global__ void __launch_bounds__(256) cvt_bf16_k(const float* __restrict__ in,
                                                  u16* __restrict__ out, int n4) {
  int i = blockIdx.x * 256 + threadIdx.x;
  if (i < n4) {
    float4 f = ((const float4*)in)[i];
    u16x4 o = { f2bf(f.x), f2bf(f.y), f2bf(f.z), f2bf(f.w) };
    ((u16x4*)out)[i] = o;
  }
}

// ---------------------------------------------------------------- NT GEMM
// C[N,M] = epilogue(A[N,K] * Bw[M,K]^T + bias[M])
// EPI: 0 = store bf16, 1 = store f32 with ReLU, 2 = store f32 plain
// GATHER: A rows indirected through map_row_local (windows without materializing)
template <int EPI, bool GATHER>
__global__ void __launch_bounds__(256) gemm_bf16_nt(
    const u16* __restrict__ A, const u16* __restrict__ Bw,
    const float* __restrict__ bias,
    u16* __restrict__ Cb, float* __restrict__ Cf,
    int N, int M, int K, int lda, int ldb, int ldc) {
  const int lane = threadIdx.x & 31;
  const int wv   = threadIdx.x >> 5;
  const int g = lane >> 4, ln = lane & 15;
  const int n0 = blockIdx.x * 128 + (wv & 3) * 32;   // 4 waves along N
  const int m0 = blockIdx.y * 64  + (wv >> 2) * 32;  // 2 waves along M

  v8f acc[2][2];
  #pragma unroll
  for (int i = 0; i < 2; ++i)
    #pragma unroll
    for (int j = 0; j < 2; ++j) acc[i][j] = {};

  size_t arow[2], brow[2];
  #pragma unroll
  for (int i = 0; i < 2; ++i) {
    int r = n0 + i * 16 + ln;
    if (r > N - 1) r = N - 1;                        // clamp; stores are guarded
    arow[i] = (GATHER ? map_row_local(r) : (size_t)r) * (size_t)lda;
  }
  #pragma unroll
  for (int j = 0; j < 2; ++j)
    brow[j] = (size_t)(m0 + j * 16 + ln) * (size_t)ldb;

  for (int k0 = 0; k0 < K; k0 += 32) {
    Frag16 af[2], bfr[2];
    #pragma unroll
    for (int i = 0; i < 2; ++i) {                    // A: 16x32, contiguous k = 16*g + s
      const u16* p = A + arow[i] + k0;
      af[i].q[0] = *(const u32x4*)(p + g * 16);
      af[i].q[1] = *(const u32x4*)(p + g * 16 + 8);
    }
    #pragma unroll
    for (int j = 0; j < 2; ++j) {                    // B: 32x16, contiguous k = 16*g + s
      const u16* p = Bw + brow[j] + k0 + g * 16;
      bfr[j].q[0] = *(const u32x4*)(p);
      bfr[j].q[1] = *(const u32x4*)(p + 8);
    }
    #pragma unroll
    for (int i = 0; i < 2; ++i)
      #pragma unroll
      for (int j = 0; j < 2; ++j)
        acc[i][j] = wmma_bf16(af[i], bfr[j], acc[i][j]);
  }

  #pragma unroll
  for (int j = 0; j < 2; ++j) {
    const int m = m0 + j * 16 + ln;
    const float bj = bias[m];
    #pragma unroll
    for (int i = 0; i < 2; ++i) {
      #pragma unroll
      for (int r = 0; r < 8; ++r) {
        const int n = n0 + i * 16 + r + 8 * g;       // C/D layout: lane-half rows
        if (n >= N) continue;
        float v = acc[i][j][r] + bj;
        if (EPI == 0)      Cb[(size_t)n * ldc + m] = f2bf(v);
        else if (EPI == 1) Cf[(size_t)n * ldc + m] = fmaxf(v, 0.0f);
        else               Cf[(size_t)n * ldc + m] = v;
      }
    }
  }
}

// ---------------------------------------------------------------- global attention
// one wave per (b, head, 16-query tile); flash-style online softmax over 512 keys.
// V tiles (32 keys x 64 dims, 4KB) are DMA'd to LDS with the CDNA5 async engine
// so the transpose reads hit LDS instead of strided global memory.
__global__ void __launch_bounds__(256) attn_global_k(const u16* __restrict__ qkv,
                                                     u16* __restrict__ ctx) {
  __shared__ u32x4 pshare[8][64];                    // 1KB P-tile per wave
  __shared__ u32x4 vshare[8][256];                   // 4KB V-tile per wave
  const int lane = threadIdx.x & 31, wv = threadIdx.x >> 5;
  const int g = lane >> 4, ln = lane & 15;
  const int wid = blockIdx.x * 8 + wv;               // 8192 waves total
  const int qt = wid & 31;
  const int h  = (wid >> 5) & 7;
  const int b  = wid >> 8;
  const size_t rowb = (size_t)b * Sn;
  u16* pbw  = (u16*)pshare[wv];
  u16* vbuf = (u16*)vshare[wv];

  Frag16 qa[2];                                      // Q[16 x 64] once
  {
    const u16* qp = qkv + (rowb + qt * 16 + ln) * 1536 + h * 64;
    #pragma unroll
    for (int s = 0; s < 2; ++s) {
      qa[s].q[0] = *(const u32x4*)(qp + s * 32 + g * 16);
      qa[s].q[1] = *(const u32x4*)(qp + s * 32 + g * 16 + 8);
    }
  }

  v8f vd[4]; float m8[8], l8[8];
  #pragma unroll
  for (int d = 0; d < 4; ++d) vd[d] = {};
  #pragma unroll
  for (int r = 0; r < 8; ++r) { m8[r] = -3.0e38f; l8[r] = 0.0f; }

  for (int j = 0; j < Sn; j += 32) {                 // 32 keys / iteration
    // make sure last iteration's LDS reads drained before async engine rewrites
    asm volatile("s_wait_dscnt 0x0" ::: "memory");
    // ---- stage V tile [32 keys x 64 dims] -> LDS (8 x b128 per lane, coalesced)
    #pragma unroll
    for (int t = 0; t < 8; ++t) {
      const int ci  = t * 32 + lane;                 // chunk id: 256 x 16B
      const int key = ci >> 3, dc = ci & 7;
      const u16* gp = qkv + (rowb + j + key) * 1536 + 1024 + h * 64 + dc * 8;
      u16* lp = vbuf + key * 64 + dc * 8;
#if HAVE_ASYNC_LDS
      __builtin_amdgcn_global_load_async_to_lds_b128((gv4i*)gp, (sv4i*)lp, 0, 0);
#else
      *(u32x4*)lp = *(const u32x4*)gp;               // VGPR bounce fallback
#endif
    }

    // ---- scores: two 16-key tiles, dh=64 -> 2 K-steps each
    v8f s0 = {}, s1 = {};
    #pragma unroll
    for (int t = 0; t < 2; ++t) {
      v8f sa = {};
      const u16* kp = qkv + (rowb + j + t * 16 + ln) * 1536 + 512 + h * 64;
      #pragma unroll
      for (int s = 0; s < 2; ++s) {
        Frag16 kb;
        kb.q[0] = *(const u32x4*)(kp + s * 32 + g * 16);
        kb.q[1] = *(const u32x4*)(kp + s * 32 + g * 16 + 8);
        sa = wmma_bf16(qa[s], kb, sa);
      }
      if (t == 0) s0 = sa; else s1 = sa;
    }
    #pragma unroll
    for (int r = 0; r < 8; ++r) {                    // online softmax per row
      float a = s0[r] * 0.125f;                      // 1/sqrt(64)
      float c = s1[r] * 0.125f;
      float mx = fmaxf(a, c);
      #pragma unroll
      for (int d = 1; d < 16; d <<= 1) mx = fmaxf(mx, __shfl_xor(mx, d, 32));
      float mn = fmaxf(m8[r], mx);
      float al = __expf(m8[r] - mn);
      float e0 = __expf(a - mn), e1 = __expf(c - mn);
      float rs = e0 + e1;
      #pragma unroll
      for (int d = 1; d < 16; d <<= 1) rs += __shfl_xor(rs, d, 32);
      l8[r] = l8[r] * al + rs;
      m8[r] = mn;
      #pragma unroll
      for (int dg = 0; dg < 4; ++dg) vd[dg][r] *= al;
      const int row = r + 8 * g;
      pbw[row * 32 + ln]      = f2bf(e0);            // P -> LDS (C-layout write)
      pbw[row * 32 + 16 + ln] = f2bf(e1);
    }
    asm volatile("s_wait_dscnt 0x0" ::: "memory");   // P visible across lanes
    Frag16 pa;                                       // read back as A-fragment
    {
      const u16* pr = pbw + ln * 32;
      pa.q[0] = *(const u32x4*)(pr + g * 16);
      pa.q[1] = *(const u32x4*)(pr + g * 16 + 8);
    }
    async_wait0();                                   // V tile resident in LDS
    #pragma unroll
    for (int dg = 0; dg < 4; ++dg) {                 // ctx += P(16x32) * V(32x16)
      Frag16 vb;                                     // transpose-read from LDS
      const u16* vp = vbuf + (g * 16) * 64 + dg * 16 + ln;
      #pragma unroll
      for (int i = 0; i < 16; ++i) vb.u[i] = vp[i * 64];
      vd[dg] = wmma_bf16(pa, vb, vd[dg]);
    }
  }
  #pragma unroll
  for (int dg = 0; dg < 4; ++dg)
    #pragma unroll
    for (int r = 0; r < 8; ++r) {
      const int row = r + 8 * g;
      ctx[(rowb + qt * 16 + row) * 512 + h * 64 + dg * 16 + ln] = f2bf(vd[dg][r] / l8[r]);
    }
}

// ---------------------------------------------------------------- local attention
// one wave per (b, window, head); W=10 padded into a 16x16 tile, key-masked softmax
__global__ void __launch_bounds__(256) attn_local_k(const u16* __restrict__ qkv,
                                                    u16* __restrict__ ctx) {
  __shared__ u32x4 pshare[8][64];
  const int lane = threadIdx.x & 31, wv = threadIdx.x >> 5;
  const int g = lane >> 4, ln = lane & 15;
  const int wid = blockIdx.x * 8 + wv;               // 13184 waves total
  const int h  = wid & 3;
  const int bn = wid >> 2;                           // b*103 + nw
  const int nw = bn % NWn;
  const size_t base = (size_t)bn * Wn;               // first qkv row of the window
  u16* pbw = (u16*)pshare[wv];

  const int idx = (ln < Wn) ? ln : (Wn - 1);         // clamp pad rows/cols
  Frag16 qa[4];
  {
    const u16* qp = qkv + (base + idx) * 1536 + h * 128;
    #pragma unroll
    for (int s = 0; s < 4; ++s) {                    // dh=128 -> 4 K-steps
      qa[s].q[0] = *(const u32x4*)(qp + s * 32 + g * 16);
      qa[s].q[1] = *(const u32x4*)(qp + s * 32 + g * 16 + 8);
    }
  }
  v8f sc = {};
  {
    const u16* kp = qkv + (base + idx) * 1536 + 512 + h * 128;
    #pragma unroll
    for (int s = 0; s < 4; ++s) {
      Frag16 kb;
      kb.q[0] = *(const u32x4*)(kp + s * 32 + g * 16);
      kb.q[1] = *(const u32x4*)(kp + s * 32 + g * 16 + 8);
      sc = wmma_bf16(qa[s], kb, sc);
    }
  }
  const bool kvalid = (ln < Wn) && (nw * STRIDEn + ln < Sn);
  float p[8];
  #pragma unroll
  for (int r = 0; r < 8; ++r) {                      // full softmax, one tile
    float v = kvalid ? sc[r] * 0.08838834764831845f : -1.0e9f;  // 1/sqrt(128)
    float mx = v;
    #pragma unroll
    for (int d = 1; d < 16; d <<= 1) mx = fmaxf(mx, __shfl_xor(mx, d, 32));
    float e = __expf(v - mx);
    float rs = e;
    #pragma unroll
    for (int d = 1; d < 16; d <<= 1) rs += __shfl_xor(rs, d, 32);
    p[r] = e / rs;
  }
  #pragma unroll
  for (int i = 0; i < 16; ++i) pbw[i * 32 + lane] = 0;       // zero-pad K=10..31
  #pragma unroll
  for (int r = 0; r < 8; ++r) pbw[(r + 8 * g) * 32 + ln] = f2bf(p[r]);
  asm volatile("s_wait_dscnt 0x0" ::: "memory");
  Frag16 pa;
  {
    const u16* pr = pbw + ln * 32;
    pa.q[0] = *(const u32x4*)(pr + g * 16);
    pa.q[1] = *(const u32x4*)(pr + g * 16 + 8);
  }
  #pragma unroll
  for (int dg = 0; dg < 8; ++dg) {                   // ctx[16x128]: single K-step
    Frag16 vb;
    const int d = dg * 16 + ln;
    #pragma unroll
    for (int i = 0; i < 16; ++i) {
      const int k = g * 16 + i;
      vb.u[i] = (k < Wn) ? qkv[(base + k) * 1536 + 1024 + h * 128 + d] : (u16)0;
    }
    v8f o = {};
    o = wmma_bf16(pa, vb, o);
    #pragma unroll
    for (int r = 0; r < 8; ++r) {
      const int w = r + 8 * g;
      if (w < Wn) ctx[(base + w) * 512 + h * 128 + d] = f2bf(o[r]);
    }
  }
}

// ---------------------------------------------------------------- local combine
// deterministic gather of the (<=2) windows covering each position; writes bf16
// into the second half of the combined activation buffer.
__global__ void __launch_bounds__(256) combine_local_k(const float* __restrict__ lattn,
                                                       u16* __restrict__ cb) {
  const int i = blockIdx.x * 256 + threadIdx.x;      // over 16384*512
  if (i >= Bn * Sn * Dn) return;
  const int m   = i & 511;
  const int n   = i >> 9;
  const int b   = n >> 9;
  const int pos = n & 511;
  float s = 0.0f;
  int lo = (pos - (Wn - 1) + (STRIDEn - 1)) / STRIDEn;   // ceil((pos-9)/5)
  if (lo < 0) lo = 0;
  int hi = pos / STRIDEn;
  if (hi > NWn - 1) hi = NWn - 1;
  for (int nw = lo; nw <= hi; ++nw) {
    const int w = pos - STRIDEn * nw;                // 0..9 by construction
    s += lattn[(((size_t)b * NWn + nw) * Wn + w) * 512 + m];
  }
  cb[(size_t)n * 1024 + 512 + m] = f2bf(s);
}

// ---------------------------------------------------------------- host driver
extern "C" void kernel_launch(void* const* d_in, const int* in_sizes, int n_in,
                              void* d_out, int out_size, void* d_ws, size_t ws_size,
                              hipStream_t stream) {
  (void)in_sizes; (void)n_in; (void)out_size; (void)ws_size;
  const float* x      = (const float*)d_in[0];
  const float* gw_in  = (const float*)d_in[1];
  const float* gb_in  = (const float*)d_in[2];
  const float* gw_out = (const float*)d_in[3];
  const float* gb_out = (const float*)d_in[4];
  const float* lw_in  = (const float*)d_in[5];
  const float* lb_in  = (const float*)d_in[6];
  const float* lw_out = (const float*)d_in[7];
  const float* lb_out = (const float*)d_in[8];
  const float* fw     = (const float*)d_in[9];
  const float* fb     = (const float*)d_in[10];
  float* out = (float*)d_out;

  char* ws = (char*)d_ws;
  size_t off = 0;
  auto take = [&](size_t bytes) {
    char* p = ws + off;
    off += (bytes + 255) & ~(size_t)255;
    return p;
  };
  u16* xb    = (u16*)take((size_t)16384 * 512 * 2);   // x as bf16
  u16* wgin  = (u16*)take((size_t)1536 * 512 * 2);
  u16* wgout = (u16*)take((size_t)512 * 512 * 2);
  u16* wlin  = (u16*)take((size_t)1536 * 512 * 2);
  u16* wlout = (u16*)take((size_t)512 * 512 * 2);
  u16* wfw   = (u16*)take((size_t)512 * 1024 * 2);
  char* qkvRegion = take((size_t)NLOC * 1536 * 2);    // 101 MB, reused 3x
  u16*   qkv   = (u16*)qkvRegion;                     // global qkv, then local qkv
  float* lattn = (float*)qkvRegion;                   // then local out-proj (f32)
  u16* ctx = (u16*)take((size_t)NLOC * 512 * 2);      // global ctx, then local ctx
  u16* cb  = (u16*)take((size_t)16384 * 1024 * 2);    // [global_out | local_out] bf16

  auto cvt = [&](const float* src, u16* dst, int n) {
    int n4 = n / 4;
    cvt_bf16_k<<<(n4 + 255) / 256, 256, 0, stream>>>(src, dst, n4);
  };
  cvt(x,      xb,    16384 * 512);
  cvt(gw_in,  wgin,  1536 * 512);
  cvt(gw_out, wgout, 512 * 512);
  cvt(lw_in,  wlin,  1536 * 512);
  cvt(lw_out, wlout, 512 * 512);
  cvt(fw,     wfw,   512 * 1024);

  // ---- global path
  gemm_bf16_nt<0, false><<<dim3(128, 24), 256, 0, stream>>>(
      xb, wgin, gb_in, qkv, nullptr, 16384, 1536, 512, 512, 512, 1536);
  attn_global_k<<<1024, 256, 0, stream>>>(qkv, ctx);
  gemm_bf16_nt<0, false><<<dim3(128, 8), 256, 0, stream>>>(
      ctx, wgout, gb_out, cb, nullptr, 16384, 512, 512, 512, 512, 1024);

  // ---- local path (window gather fused into the QKV GEMM's A-row indexing)
  gemm_bf16_nt<0, true><<<dim3(258, 24), 256, 0, stream>>>(
      xb, wlin, lb_in, qkv, nullptr, NLOC, 1536, 512, 512, 512, 1536);
  attn_local_k<<<1648, 256, 0, stream>>>(qkv, ctx);
  gemm_bf16_nt<2, false><<<dim3(258, 8), 256, 0, stream>>>(
      ctx, wlout, lb_out, nullptr, lattn, NLOC, 512, 512, 512, 512, 512);
  combine_local_k<<<(16384 * 512 + 255) / 256, 256, 0, stream>>>(lattn, cb);

  // ---- final fused FFN: relu([g|l] @ fw^T + fb)
  gemm_bf16_nt<1, false><<<dim3(128, 8), 256, 0, stream>>>(
      cb, wfw, fb, nullptr, out, 16384, 512, 1024, 1024, 1024, 512);
}
